// PalaceAttention_67534065762979
// MI455X (gfx1250) — matrix operand order repaired and checked by reading, not compiled, not run on hardware
//
#include <hip/hip_runtime.h>

// ---------- types ----------
typedef __attribute__((ext_vector_type(16))) __bf16 v16bf;
typedef __attribute__((ext_vector_type(8)))  float  v8f;

struct Frag32 { uint4 lo, hi; };   // 32 bytes = 16 bf16 elements (8 VGPRs)

static __device__ __forceinline__ v16bf frag_cast(Frag32 f) {
    return __builtin_bit_cast(v16bf, f);
}

// f32 -> bf16, round-to-nearest-even
static __device__ __forceinline__ unsigned short f2bf(float f) {
    unsigned int u = __float_as_uint(f);
    unsigned int r = (u + 0x7FFFu + ((u >> 16) & 1u)) >> 16;
    return (unsigned short)r;
}

// Dimensions (fixed by the reference)
#define BATCH 4
#define SEQ   2048
#define DIM   1024
#define HEADS 16
#define HD    64

// ---------- stage 0: f32 -> bf16 conversion ----------
__global__ void cvt_f32_bf16_x4(const float* __restrict__ src,
                                unsigned short* __restrict__ dst, int n4) {
    int i = blockIdx.x * blockDim.x + threadIdx.x;
    if (i >= n4) return;
    float4 v = ((const float4*)src)[i];
    uint2 o;
    o.x = (unsigned)f2bf(v.x) | ((unsigned)f2bf(v.y) << 16);
    o.y = (unsigned)f2bf(v.z) | ((unsigned)f2bf(v.w) << 16);
    ((uint2*)dst)[i] = o;
}

// ---------- stage 1: QKV projection GEMM ----------
// C[m][n] = sum_k x[m][k] * W[n][k]   (Q = x @ W^T)
// grid: (8192/128, 1024/64, 3), block 256 (8 waves)
// z==0 -> Q row-major bf16, z==1 -> K row-major bf16, z==2 -> V transposed bf16
__global__ __launch_bounds__(256) void qkv_gemm(
    const unsigned short* __restrict__ xb,
    const unsigned short* __restrict__ Wb,
    unsigned short* __restrict__ Qb,
    unsigned short* __restrict__ Kb,
    unsigned short* __restrict__ Vt)
{
    constexpr int LDA = 40;                    // padded row stride (ushorts)
    __shared__ unsigned short As[128 * LDA];   // 128x32 A tile
    __shared__ unsigned short Bs[64  * LDA];   // 64x32  B tile (W rows)

    const int z    = blockIdx.z;
    const unsigned short* W = Wb + (size_t)z * DIM * DIM;
    const int row0 = blockIdx.x * 128;
    const int col0 = blockIdx.y * 64;
    const int tid  = threadIdx.x;
    const int lane = tid & 31;
    const int wid  = tid >> 5;
    const int wm   = wid >> 1;                 // 0..3  (M sub-tile)
    const int wn   = wid & 1;                  // 0..1  (N sub-tile)
    const int r    = lane & 15;
    const int hf   = lane >> 4;

    v8f acc[2][2] = {};

    for (int k0 = 0; k0 < DIM; k0 += 32) {
        __syncthreads();
        // A tile: 128 rows x 32 cols = 512 x uint4 -> 2 per thread
        #pragma unroll
        for (int i = 0; i < 2; ++i) {
            int task = tid + i * 256;
            int arow = task >> 2, aseg = task & 3;
            uint4 d = *(const uint4*)(xb + (size_t)(row0 + arow) * DIM + k0 + aseg * 8);
            *(uint4*)(As + arow * LDA + aseg * 8) = d;
        }
        // B tile: 64 rows x 32 cols = 256 x uint4 -> 1 per thread
        {
            int brow = tid >> 2, bseg = tid & 3;
            uint4 d = *(const uint4*)(W + (size_t)(col0 + brow) * DIM + k0 + bseg * 8);
            *(uint4*)(Bs + brow * LDA + bseg * 8) = d;
        }
        if (k0 + 32 < DIM) {  // hint next tiles toward caches
            __builtin_prefetch(xb + (size_t)(row0 + (tid >> 1)) * DIM + k0 + 32, 0, 1);
            __builtin_prefetch(W  + (size_t)(col0 + (tid & 63)) * DIM + k0 + 32, 0, 1);
        }
        __syncthreads();

        Frag32 af[2], bf[2];
        #pragma unroll
        for (int i = 0; i < 2; ++i) {
            const unsigned short* p = As + (wm * 32 + i * 16 + r) * LDA;
            af[i].lo = *(const uint4*)(p + hf * 8);           // K = hf*8 .. +7
            af[i].hi = *(const uint4*)(p + 16 + hf * 8);      // K = 16+hf*8 ..
        }
        #pragma unroll
        for (int j = 0; j < 2; ++j) {
            const unsigned short* p = Bs + (wn * 32 + j * 16 + r) * LDA;
            bf[j].lo = *(const uint4*)(p + hf * 8);           // K = hf*8 .. +7
            bf[j].hi = *(const uint4*)(p + 16 + hf * 8);      // K = 16+hf*8 ..
        }
        #pragma unroll
        for (int i = 0; i < 2; ++i)
            #pragma unroll
            for (int j = 0; j < 2; ++j)
                acc[i][j] = __builtin_amdgcn_wmma_f32_16x16x32_bf16(
                    false, frag_cast(af[i]), false, frag_cast(bf[j]),
                    (short)0, acc[i][j], false, false);
    }

    // C/D layout: VGPR g -> row g + 8*hf, col = r
    #pragma unroll
    for (int i = 0; i < 2; ++i)
        #pragma unroll
        for (int j = 0; j < 2; ++j)
            #pragma unroll
            for (int g = 0; g < 8; ++g) {
                int m = row0 + wm * 32 + i * 16 + g + hf * 8;
                int n = col0 + wn * 32 + j * 16 + r;
                unsigned short hv = f2bf(acc[i][j][g]);
                if (z == 0)      Qb[(size_t)m * DIM + n] = hv;
                else if (z == 1) Kb[(size_t)m * DIM + n] = hv;
                else {
                    int bb = m >> 11, t = m & (SEQ - 1);
                    Vt[((size_t)bb * DIM + n) * SEQ + t] = hv;  // transposed V
                }
            }
}

// ---------- stage 2: masked flash attention (all-transposed, 64-wide k blocks) ----------
// S^T = K Q^T  (q = C-layout column -> per-lane scalar softmax stats)
// O^T = V^T P^T (V^T staged row-major as Vt; P^T re-laid as B frags via shuffles)
// grid: (SEQ/64, HEADS, BATCH), block 128 (4 waves, 16 q-rows per wave)
__global__ __launch_bounds__(128) void palace_attn(
    const unsigned short* __restrict__ Qb,
    const unsigned short* __restrict__ Kb,
    const unsigned short* __restrict__ Vt,
    const float* __restrict__ wptr,
    float* __restrict__ out)
{
    const int qb = blockIdx.x, h = blockIdx.y, b = blockIdx.z;
    const int tid  = threadIdx.x;
    const int lane = tid & 31, wid = tid >> 5;
    const int r = lane & 15, hf = lane >> 4;
    const int q0 = qb * 64 + wid * 16;

    const float sigw  = 1.0f / (1.0f + __expf(-wptr[0]));
    const float scale = 0.125f;                     // hd^-0.5 = 1/8

    // persistent Q^T fragments (B operand): lane = q column, K-striped d
    Frag32 bq[2];
    {
        const unsigned short* qp = Qb + (size_t)(b * SEQ + q0 + r) * DIM + h * HD;
        #pragma unroll
        for (int c = 0; c < 2; ++c) {
            bq[c].lo = *(const uint4*)(qp + c * 32 + hf * 8);
            bq[c].hi = *(const uint4*)(qp + c * 32 + 16 + hf * 8);
        }
    }

    v8f accO[4] = {};                 // O^T: row d = t2*16 + g + 8*hf, col q = r
    float mrow = -3.0e38f, lrow = 0.0f;   // per-lane scalars (q = q0 + r)
    const int qpid = ((q0 + r) & 63) >> 3;

    for (int kb = 0; kb < SEQ / 64; ++kb) {
        // ---- S^T tiles: t in {0..3}, each 16(k) x 16(q); 4 independent chains
        v8f s[4];
        #pragma unroll
        for (int t = 0; t < 4; ++t) {
            const unsigned short* kp =
                Kb + (size_t)(b * SEQ + kb * 64 + t * 16 + r) * DIM + h * HD;
            Frag32 ak0, ak1;                              // A operand: K rows
            ak0.lo = *(const uint4*)(kp + hf * 8);
            ak0.hi = *(const uint4*)(kp + 16 + hf * 8);
            ak1.lo = *(const uint4*)(kp + 32 + hf * 8);
            ak1.hi = *(const uint4*)(kp + 32 + 16 + hf * 8);
            v8f zc = {};
            zc = __builtin_amdgcn_wmma_f32_16x16x32_bf16(
                false, frag_cast(ak0), false, frag_cast(bq[0]), (short)0, zc, false, false);
            s[t] = __builtin_amdgcn_wmma_f32_16x16x32_bf16(
                false, frag_cast(ak1), false, frag_cast(bq[1]), (short)0, zc, false, false);
        }
        if (kb + 1 < SEQ / 64) {
            __builtin_prefetch(Kb + (size_t)(b * SEQ + (kb + 1) * 64 + lane) * DIM + h * HD, 0, 1);
            __builtin_prefetch(Kb + (size_t)(b * SEQ + (kb + 1) * 64 + 32 + lane) * DIM + h * HD, 0, 1);
        }

        // ---- scale + multiplicative palace mask; k = kb*64 + t*16 + g + 8*hf
        float pv[4][8];                    // holds sv, then exp(sv - m) in place
        #pragma unroll
        for (int t = 0; t < 4; ++t)
            #pragma unroll
            for (int g = 0; g < 8; ++g) {
                int kk  = kb * 64 + t * 16 + g + hf * 8;
                int kpd = (kk & 63) >> 3;
                pv[t][g] = s[t][g] * scale * ((qpid == kpd) ? 1.0f : sigw);
            }

        // ---- online softmax: per-lane reduce over 32 regs + 1 cross-half shuffle
        float mx = pv[0][0];
        #pragma unroll
        for (int t = 0; t < 4; ++t)
            #pragma unroll
            for (int g = 0; g < 8; ++g) mx = fmaxf(mx, pv[t][g]);
        mx = fmaxf(mx, __shfl_xor(mx, 16, 32));
        float mnew  = fmaxf(mrow, mx);
        float alpha = __expf(mrow - mnew);
        mrow = mnew;

        float sum = 0.0f;
        #pragma unroll
        for (int t = 0; t < 4; ++t)
            #pragma unroll
            for (int g = 0; g < 8; ++g) {
                pv[t][g] = __expf(pv[t][g] - mnew);
                sum += pv[t][g];
            }
        sum += __shfl_xor(sum, 16, 32);
        lrow = lrow * alpha + sum;
        #pragma unroll
        for (int t2 = 0; t2 < 4; ++t2)
            #pragma unroll
            for (int g = 0; g < 8; ++g) accO[t2][g] *= alpha;

        // ---- P^T -> two B fragments (k 0..31 and 32..63), register-only exchange:
        // within each 32-block (tiles 2p, 2p+1), lane(hf) holds k in
        // {8hf..8hf+7} U {16+8hf..}; B frags here use {16hf..16hf+15}, matching the V^T A frags.
        Frag32 pb[2];
        #pragma unroll
        for (int p = 0; p < 2; ++p) {
            float ex[8], ownv[8];
            #pragma unroll
            for (int g = 0; g < 8; ++g) {
                float snd = hf ? pv[2 * p][g] : pv[2 * p + 1][g];
                ex[g]   = __shfl_xor(snd, 16, 32);
                ownv[g] = hf ? pv[2 * p + 1][g] : pv[2 * p][g];
            }
            unsigned pk[8];
            #pragma unroll
            for (int j = 0; j < 4; ++j) {
                float l0 = hf ? ex[2 * j]     : ownv[2 * j];
                float l1 = hf ? ex[2 * j + 1] : ownv[2 * j + 1];
                float h0 = hf ? ownv[2 * j]     : ex[2 * j];
                float h1 = hf ? ownv[2 * j + 1] : ex[2 * j + 1];
                pk[j]     = (unsigned)f2bf(l0) | ((unsigned)f2bf(l1) << 16);
                pk[4 + j] = (unsigned)f2bf(h0) | ((unsigned)f2bf(h1) << 16);
            }
            pb[p].lo = make_uint4(pk[0], pk[1], pk[2], pk[3]);
            pb[p].hi = make_uint4(pk[4], pk[5], pk[6], pk[7]);
        }

        // ---- O^T += V^T x P^T   (V^T rows are contiguous A fragments)
        #pragma unroll
        for (int t2 = 0; t2 < 4; ++t2) {
            const unsigned short* vp =
                Vt + (size_t)(b * DIM + h * HD + t2 * 16 + r) * SEQ + kb * 64;
            #pragma unroll
            for (int p = 0; p < 2; ++p) {
                Frag32 av;
                av.lo = *(const uint4*)(vp + p * 32 + hf * 16);
                av.hi = *(const uint4*)(vp + p * 32 + hf * 16 + 8);
                accO[t2] = __builtin_amdgcn_wmma_f32_16x16x32_bf16(
                    false, frag_cast(av), false, frag_cast(pb[p]), (short)0, accO[t2], false, false);
            }
        }
    }

    // ---- normalize (scalar per lane) + vectorized store:
    // out[b][q0+r][h*64 + t2*16 + 8*hf + g], g=0..7 contiguous -> 2x float4 per tile
    float inv = 1.0f / lrow;
    float* orow = out + (size_t)(b * SEQ + q0 + r) * DIM + h * HD + hf * 8;
    #pragma unroll
    for (int t2 = 0; t2 < 4; ++t2) {
        float4 lo4 = make_float4(accO[t2][0] * inv, accO[t2][1] * inv,
                                 accO[t2][2] * inv, accO[t2][3] * inv);
        float4 hi4 = make_float4(accO[t2][4] * inv, accO[t2][5] * inv,
                                 accO[t2][6] * inv, accO[t2][7] * inv);
        *(float4*)(orow + t2 * 16 + 0) = lo4;
        *(float4*)(orow + t2 * 16 + 4) = hi4;
    }
}

// ---------- launcher ----------
extern "C" void kernel_launch(void* const* d_in, const int* in_sizes, int n_in,
                              void* d_out, int out_size, void* d_ws, size_t ws_size,
                              hipStream_t stream) {
    const float* x  = (const float*)d_in[0];
    const float* Wq = (const float*)d_in[1];
    const float* Wk = (const float*)d_in[2];
    const float* Wv = (const float*)d_in[3];
    const float* w  = (const float*)d_in[4];

    // workspace carve (bf16 ushorts): x | W(q,k,v) | Q | K | Vt  = ~70 MB
    unsigned short* xb = (unsigned short*)d_ws;
    unsigned short* Wb = xb + (size_t)BATCH * SEQ * DIM;        // 8M
    unsigned short* Qb = Wb + (size_t)3 * DIM * DIM;            // +3M
    unsigned short* Kb = Qb + (size_t)BATCH * SEQ * DIM;        // +8M
    unsigned short* Vt = Kb + (size_t)BATCH * SEQ * DIM;        // +8M

    // stage 0: conversions
    {
        int n4 = (BATCH * SEQ * DIM) / 4;
        cvt_f32_bf16_x4<<<(n4 + 255) / 256, 256, 0, stream>>>(x, xb, n4);
        int w4 = (DIM * DIM) / 4;
        cvt_f32_bf16_x4<<<(w4 + 255) / 256, 256, 0, stream>>>(Wq, Wb, w4);
        cvt_f32_bf16_x4<<<(w4 + 255) / 256, 256, 0, stream>>>(Wk, Wb + (size_t)DIM * DIM, w4);
        cvt_f32_bf16_x4<<<(w4 + 255) / 256, 256, 0, stream>>>(Wv, Wb + (size_t)2 * DIM * DIM, w4);
    }

    // stage 1: QKV projections
    qkv_gemm<<<dim3((BATCH * SEQ) / 128, DIM / 64, 3), 256, 0, stream>>>(xb, Wb, Qb, Kb, Vt);

    // stage 2: masked attention
    palace_attn<<<dim3(SEQ / 64, HEADS, BATCH), 128, 0, stream>>>(Qb, Kb, Vt, w, (float*)d_out);
}
